// InterventionDissimilarityLoss_38096359916255
// MI455X (gfx1250) — compile-verified, matrix operand order checked by reading, not run on hardware
//
#include <hip/hip_runtime.h>

// ---------------------------------------------------------------------------
// InterventionDissimilarityLoss for MI455X (gfx1250, wave32).
//
// Memory-bound reduction: ~192MB (64MB pred_int + 128MB gt_mask) @ 23.3TB/s
// -> ~8us floor. Strategy:
//   pass1: grid (tiles=32, b=32), 256 threads. Per block:
//     - read GT rows coalesced ONCE (int4), compress into 64-bit/t k-bitmask
//       in LDS via order-independent ds_or_b32 atomics (deterministic),
//     - stream all 64 k-slices of pred_intervened coalesced,
//     - per-k 32-lane reduction via v_wmma_f32_16x16x32_f16 with B=ones
//       (hi/lo f16 split keeps f32-level accuracy; counts exact),
//     - write float4 partial (sumA,sumN,cntA,cntN) per (k,b,tile) to d_ws.
//   pass2: 1 block, fixed-order deterministic combine -> 2 scalar outputs.
// ---------------------------------------------------------------------------

#define B_      32
#define TV      8192
#define K_      64
#define TS      128
#define TT      256              // t-values per block (one per thread)
#define NTILES  (TV / TT)        // 32
#define MARGIN_ 0.1f
#define LAMBDA_ 0.5f

typedef _Float16 v16h __attribute__((ext_vector_type(16)));
typedef float    v8f  __attribute__((ext_vector_type(8)));

__device__ __forceinline__ float sigmoidf_(float x) {
    return 1.0f / (1.0f + __expf(-x));
}

// Exact-ish 32-lane sum using one WMMA.
// A (16x32 f16) layout: lane m (m<16) holds K=0,1 in VGPR0 halves; lane m+16
// holds K=8,9. With slot0=hi(x), slot1=lo(x) and B=all-ones:
//   D[m,n] = x(m) + x(m+16)   (f16-split -> ~2^-22 relative error)
// D layout: lane n<16 VGPRj = D[j,n]; lane 16+n VGPRj = D[8+j,n].
// Sum 8 D components per lane, then shfl_xor(16) -> full 32-lane sum in
// every lane. EXEC is all-ones at every call site (uniform control flow).
__device__ __forceinline__ float wave_sum32(float x) {
    v16h a = {};
    _Float16 h = (_Float16)x;
    a[0] = h;
    a[1] = (_Float16)(x - (float)h);
    v16h bv;
#pragma unroll
    for (int i = 0; i < 16; ++i) bv[i] = (_Float16)1.0f;
    v8f c = {};
    v8f d = __builtin_amdgcn_wmma_f32_16x16x32_f16(
        /*neg_a=*/false, a, /*neg_b=*/false, bv,
        /*c_mod=*/(short)0, c, /*reuse_a=*/false, /*reuse_b=*/false);
    float s = d[0] + d[1] + d[2] + d[3] + d[4] + d[5] + d[6] + d[7];
    s += __shfl_xor(s, 16, 32);
    return s;
}

__global__ __launch_bounds__(TT) void idl_pass1(
    const float* __restrict__ pred_orig,   // [B,TV]
    const float* __restrict__ pred_int,    // [K,B,TV]
    const int*   __restrict__ idxs,        // [K]
    const int*   __restrict__ gt_mask,     // [B,TV,TS]
    const int*   __restrict__ mask_video,  // [B,TV]
    float4*      __restrict__ partials)    // [K,B,NTILES]
{
    const int tile = blockIdx.x;
    const int b    = blockIdx.y;
    const int tid  = threadIdx.x;
    const int t0   = tile * TT;

    __shared__ int      sIdx[K_];
    __shared__ unsigned kmLo[TS], kmHi[TS];     // column s -> k-bit masks
    __shared__ unsigned bitsLo[TT], bitsHi[TT]; // per-t: bit k = gt[b,t,idx[k]]>0
    __shared__ float    poS[TT];                // sigmoid(pred_orig)
    __shared__ unsigned validS[TT];
    __shared__ float    red[K_][8][4];          // per-(k,wave) partials

    // Phase A: stage indices, zero bitmasks, sigmoid(pred_orig), validity.
    if (tid < K_) sIdx[tid] = idxs[tid];
    bitsLo[tid] = 0u;
    bitsHi[tid] = 0u;
    {
        const size_t off = (size_t)b * TV + t0 + tid;
        poS[tid]    = sigmoidf_(pred_orig[off]);
        validS[tid] = (mask_video[off] != 0) ? 1u : 0u;
    }
    __syncthreads();

    // Phase B: column -> k-bitmask map (handles duplicate indices).
    if (tid < TS) {
        unsigned lo = 0u, hi = 0u;
        for (int k = 0; k < K_; ++k) {
            if (sIdx[k] == tid) {
                if (k < 32) lo |= (1u << k);
                else        hi |= (1u << (k - 32));
            }
        }
        kmLo[tid] = lo;
        kmHi[tid] = hi;
    }
    __syncthreads();

    // Phase C: coalesced one-pass read of GT rows; compress to bitmasks.
    {
        const int* gtB = gt_mask + ((size_t)b * TV + t0) * TS;
        for (int it = 0; it < TT / 8; ++it) {
            const int r = (it << 3) + (tid >> 5);   // row within tile
            const int c = tid & 31;                 // int4 chunk within row
            const int4 g = ((const int4*)(gtB + (size_t)r * TS))[c];
            const int s0 = c << 2;
            unsigned mlo = 0u, mhi = 0u;
            if (g.x > 0) { mlo |= kmLo[s0 + 0]; mhi |= kmHi[s0 + 0]; }
            if (g.y > 0) { mlo |= kmLo[s0 + 1]; mhi |= kmHi[s0 + 1]; }
            if (g.z > 0) { mlo |= kmLo[s0 + 2]; mhi |= kmHi[s0 + 2]; }
            if (g.w > 0) { mlo |= kmLo[s0 + 3]; mhi |= kmHi[s0 + 3]; }
            if (mlo) atomicOr(&bitsLo[r], mlo);
            if (mhi) atomicOr(&bitsHi[r], mhi);
        }
    }
    __syncthreads();

    // Phase D: stream all K slices of pred_intervened (coalesced b32 loads),
    // WMMA wave-reduce per k, stash per-wave partials in LDS.
    {
        const float    ps  = poS[tid];
        const unsigned v   = validS[tid];
        const unsigned blo = bitsLo[tid];
        const unsigned bhi = bitsHi[tid];
        const float* piBase = pred_int + (size_t)b * TV + t0 + tid;
        const size_t kStride = (size_t)B_ * TV;
        const int wv = tid >> 5;

        for (int k = 0; k < K_; ++k) {
            const float x = piBase[(size_t)k * kStride];
            if (k + 1 < K_)
                __builtin_prefetch(piBase + (size_t)(k + 1) * kStride, 0, 0);
            const float e = fabsf(ps - sigmoidf_(x));
            const unsigned bit = (k < 32) ? ((blo >> k) & 1u)
                                          : ((bhi >> (k - 32)) & 1u);
            const float av = (v & bit)          ? 1.0f : 0.0f;
            const float nv = (v & (bit ^ 1u))   ? 1.0f : 0.0f;

            const float sa = wave_sum32(e * av);
            const float sn = wave_sum32(e * nv);
            const float ca = wave_sum32(av);
            const float cn = wave_sum32(nv);
            if ((tid & 31) == 0) {
                red[k][wv][0] = sa;
                red[k][wv][1] = sn;
                red[k][wv][2] = ca;
                red[k][wv][3] = cn;
            }
        }
    }
    __syncthreads();

    // Phase E: combine the 8 waves, write float4 partial per (k,b,tile).
    if (tid < K_) {
        float a0 = 0.f, a1 = 0.f, a2 = 0.f, a3 = 0.f;
#pragma unroll
        for (int w = 0; w < 8; ++w) {
            a0 += red[tid][w][0];
            a1 += red[tid][w][1];
            a2 += red[tid][w][2];
            a3 += red[tid][w][3];
        }
        float4 out;
        out.x = a0; out.y = a1; out.z = a2; out.w = a3;
        partials[((size_t)tid * B_ + b) * NTILES + tile] = out;
    }
}

__global__ __launch_bounds__(256) void idl_pass2(
    const float4* __restrict__ partials,   // [K,B,NTILES]
    float*        __restrict__ out)        // [2]
{
    const int tid = threadIdx.x;
    float pp = 0.0f;
    for (int kb = tid; kb < K_ * B_; kb += 256) {
        const float4* p = partials + (size_t)kb * NTILES;
        float sa = 0.f, sn = 0.f, ca = 0.f, cn = 0.f;
        for (int i = 0; i < NTILES; ++i) {   // fixed order -> deterministic
            const float4 q = p[i];
            sa += q.x; sn += q.y; ca += q.z; cn += q.w;
        }
        const float ea = sa / fmaxf(ca, 1.0f);
        const float en = sn / fmaxf(cn, 1.0f);
        if (ca > 0.0f && cn > 0.0f)
            pp += fmaxf(0.0f, MARGIN_ - (ea - en));
    }
    __shared__ float r[256];
    r[tid] = pp;
    __syncthreads();
    for (int s = 128; s > 0; s >>= 1) {
        if (tid < s) r[tid] += r[tid + s];
        __syncthreads();
    }
    if (tid == 0) {
        const float ac = r[0] / (float)(K_ * B_);
        out[0] = ac;             // alignment_consistency
        out[1] = ac * LAMBDA_;   // total_intervention_loss
    }
}

extern "C" void kernel_launch(void* const* d_in, const int* in_sizes, int n_in,
                              void* d_out, int out_size, void* d_ws, size_t ws_size,
                              hipStream_t stream) {
    const float* pred_orig = (const float*)d_in[0];   // [B,TV] f32
    const float* pred_int  = (const float*)d_in[1];   // [K,B,TV] f32
    const int*   idxs      = (const int*)d_in[2];     // [K] i32
    const int*   gt        = (const int*)d_in[3];     // [B,TV,TS] i32
    const int*   mv        = (const int*)d_in[4];     // [B,TV] i32
    float4* partials = (float4*)d_ws;                 // K*B*NTILES*16B = 1MB

    idl_pass1<<<dim3(NTILES, B_), TT, 0, stream>>>(
        pred_orig, pred_int, idxs, gt, mv, partials);
    idl_pass2<<<1, 256, 0, stream>>>(partials, (float*)d_out);
}
